// PointNet2Detector_32865089749524
// MI455X (gfx1250) — matrix operand order repaired and checked by reading, not compiled, not run on hardware
//
#include <hip/hip_runtime.h>
#include <hip/hip_bf16.h>

// ---------------------------------------------------------------------------
// PointNet++ detector forward for MI455X (gfx1250).
//
//  * Channel-MLPs are GEMMs run on v_wmma_f32_16x16x32_f16 (f16 operands,
//    f32 accumulate). Activations are kept f16 END-TO-END (largest buffer
//    262144x128 -> 67MB, L2-resident) so GEMM staging is a pure copy.
//  * Tile staging uses CDNA5 async global->LDS copies
//    (global_load_async_to_lds_b128, ASYNCcnt), DOUBLE BUFFERED: tile k+1
//    streams into LDS while tile k computes. Async loads complete in order,
//    so s_wait_asynccnt 0x2 (leave only the newest A+B pair outstanding)
//    guarantees the current tile has landed.
//  * No bounds guards in the GEMM hot loop: K padded to 32 (zero fill at
//    tensor construction / weight conversion), M%64==0, Cout%64==0 at every
//    call site. Block tile 64x64, 8 waves (4Mx2N), 2 wmma/wave/K-step.
// ---------------------------------------------------------------------------

typedef __attribute__((ext_vector_type(16))) _Float16 v16h;
typedef __attribute__((ext_vector_type(8)))  _Float16 v8h;
typedef __attribute__((ext_vector_type(8)))  float    v8f;

#define THREADS 256

// Low 32 bits of a generic shared pointer = wave-relative LDS byte offset
// (AS3 -> generic addrspacecast puts the LDS offset in addr[31:0]).
__device__ __forceinline__ unsigned lds_offset(const void* p) {
  return (unsigned)(uintptr_t)p;
}

__device__ __forceinline__ void async_copy_b128(unsigned lds, const void* gaddr) {
  asm volatile("global_load_async_to_lds_b128 %0, %1, off"
               :: "v"(lds), "v"((unsigned long long)(uintptr_t)gaddr)
               : "memory");
}

// ---------------------------------------------------------------------------
// Weight pre-conversion: W[Cout x Kreal] f32 -> Wh[Cout x Kp] f16, zero pad.
// ---------------------------------------------------------------------------
__global__ void k_w2h(const float* __restrict__ W, int Cout, int Kreal, int Kp,
                      _Float16* __restrict__ out) {
  int t = blockIdx.x * blockDim.x + threadIdx.x;
  if (t >= Cout * Kp) return;
  int o = t / Kp, k = t % Kp;
  out[t] = (k < Kreal) ? (_Float16)W[(size_t)o * Kreal + k] : (_Float16)0.0f;
}

// ---------------------------------------------------------------------------
// WMMA GEMM: out[M x Cout](f16) = A[M x Kp](f16) * Wh[Cout x Kp]^T + bias.
// Preconditions: Kp%32==0, M%64==0, Cout%64==0, rows stride = Kp, zero padded.
// ---------------------------------------------------------------------------
__global__ __launch_bounds__(THREADS)
void k_gemm_wmma_f16(const _Float16* __restrict__ A, int M, int Kp,
                     const _Float16* __restrict__ Wh, const float* __restrict__ bias,
                     _Float16* __restrict__ out, int Cout, int act) {
  __shared__ __align__(16) _Float16 sA[2][64 * 32];   // [buf][row][k]
  __shared__ __align__(16) _Float16 sB[2][64 * 32];   // [buf][n][k]

  const int lane = threadIdx.x & 31;
  const int wave = threadIdx.x >> 5;
  const int wm = wave & 3;        // M tile within block (0..3)
  const int wn = wave >> 2;       // N pair within block (0..1)
  const int row_base = blockIdx.y * 64;
  const int col_base = blockIdx.x * 64;

  // Staging: thread -> (row sr, 8-half k chunk sk); 64 rows x 4 chunks = 256.
  const int sr = threadIdx.x >> 2;
  const int sk = (threadIdx.x & 3) << 3;
  const _Float16* aRow = A  + (size_t)(row_base + sr) * Kp + sk;
  const _Float16* bRow = Wh + (size_t)(col_base + sr) * Kp + sk;
  const unsigned ldsA[2] = { lds_offset(&sA[0][sr * 32 + sk]),
                             lds_offset(&sA[1][sr * 32 + sk]) };
  const unsigned ldsB[2] = { lds_offset(&sB[0][sr * 32 + sk]),
                             lds_offset(&sB[1][sr * 32 + sk]) };

  v8f acc0 = {}, acc1 = {};
  const int ktiles = Kp >> 5;

  // Prologue: stream tile 0 into buffer 0.
  async_copy_b128(ldsA[0], aRow);
  async_copy_b128(ldsB[0], bRow);

  for (int kt = 0; kt < ktiles; ++kt) {
    const int buf = kt & 1;
    if (kt + 1 < ktiles) {
      // Stream next tile into the other buffer while this one computes.
      async_copy_b128(ldsA[buf ^ 1], aRow + (kt + 1) * 32);
      async_copy_b128(ldsB[buf ^ 1], bRow + (kt + 1) * 32);
      if (kt + 2 < ktiles) __builtin_prefetch(aRow + (kt + 2) * 32, 0, 1);
      // Async loads retire in order: <=2 outstanding == current tile landed.
      asm volatile("s_wait_asynccnt 0x2" ::: "memory");
    } else {
      asm volatile("s_wait_asynccnt 0x0" ::: "memory");
    }
    __syncthreads();

    // Fragments use contiguous k order (k = 16*hb + s), the same for A and B.
    const int hb = lane >> 4, lh = lane & 15;
    // A 16x32: lanes 0-15 (row=lh) hold K 0..15; lanes 16-31 hold K 16..31.
    const _Float16* ap = &sA[buf][(wm * 16 + lh) * 32 + hb * 16];
    v8h a_lo = *(const v8h*)ap;
    v8h a_hi = *(const v8h*)(ap + 8);
    v16h af = __builtin_shufflevector(a_lo, a_hi,
        0, 1, 2, 3, 4, 5, 6, 7, 8, 9, 10, 11, 12, 13, 14, 15);
    // B 32x16: lanes 0-15 (col=lh) hold K 0..15; lanes 16-31 hold K 16..31.
    const _Float16* bp0 = &sB[buf][(wn * 32 + lh) * 32 + hb * 16];
    v8h b0_lo = *(const v8h*)bp0;
    v8h b0_hi = *(const v8h*)(bp0 + 8);
    v16h bf0 = __builtin_shufflevector(b0_lo, b0_hi,
        0, 1, 2, 3, 4, 5, 6, 7, 8, 9, 10, 11, 12, 13, 14, 15);
    const _Float16* bp1 = bp0 + 16 * 32;
    v8h b1_lo = *(const v8h*)bp1;
    v8h b1_hi = *(const v8h*)(bp1 + 8);
    v16h bf1 = __builtin_shufflevector(b1_lo, b1_hi,
        0, 1, 2, 3, 4, 5, 6, 7, 8, 9, 10, 11, 12, 13, 14, 15);

    acc0 = __builtin_amdgcn_wmma_f32_16x16x32_f16(
        false, af, false, bf0, (short)0, acc0, false, false);
    acc1 = __builtin_amdgcn_wmma_f32_16x16x32_f16(
        false, af, false, bf1, (short)0, acc1, false, false);
    __syncthreads();
  }

  // Epilogue: C layout VGPR r -> M = r + 8*(lane>=16), N = lane&15.
  const int hb = lane >> 4, lh = lane & 15;
  const int col0 = col_base + wn * 32 + lh;
  const int col1 = col0 + 16;
  const float bv0 = bias[col0], bv1 = bias[col1];
#pragma unroll
  for (int r = 0; r < 8; ++r) {
    int row = row_base + wm * 16 + hb * 8 + r;
    float v0 = acc0[r] + bv0;
    float v1 = acc1[r] + bv1;
    if (act) { v0 = fmaxf(v0, 0.0f); v1 = fmaxf(v1, 0.0f); }
    out[(size_t)row * Cout + col0] = (_Float16)v0;
    out[(size_t)row * Cout + col1] = (_Float16)v1;
  }
}

// ---------------------------------------------------------------------------
// Farthest point sampling: one block per batch, serial over samples.
// ---------------------------------------------------------------------------
__global__ __launch_bounds__(THREADS)
void k_fps(const float* __restrict__ xyz, int N, int S,
           float* __restrict__ dist, int* __restrict__ out) {
  const int b = blockIdx.x;
  const float* p = xyz + (size_t)b * N * 3;
  float* dd = dist + (size_t)b * N;
  int* o = out + (size_t)b * S;
  __shared__ float sred[THREADS];
  __shared__ int sidx[THREADS];
  __shared__ int cur;
  __shared__ float cx, cy, cz;

  for (int i = threadIdx.x; i < N; i += THREADS) dd[i] = 1e10f;
  if (threadIdx.x == 0) cur = 0;
  __syncthreads();

  for (int s = 0; s < S; ++s) {
    if (threadIdx.x == 0) {
      o[s] = cur;
      cx = p[cur * 3 + 0]; cy = p[cur * 3 + 1]; cz = p[cur * 3 + 2];
    }
    __syncthreads();
    float best = -1.0f; int bi = 0;
    for (int i = threadIdx.x; i < N; i += THREADS) {
      float dx = p[i * 3 + 0] - cx;
      float dy = p[i * 3 + 1] - cy;
      float dz = p[i * 3 + 2] - cz;
      float d = dx * dx + dy * dy + dz * dz;
      float m = fminf(dd[i], d);
      dd[i] = m;
      if (m > best) { best = m; bi = i; }
    }
    sred[threadIdx.x] = best; sidx[threadIdx.x] = bi;
    __syncthreads();
    for (int off = THREADS / 2; off > 0; off >>= 1) {
      if (threadIdx.x < off && sred[threadIdx.x + off] > sred[threadIdx.x]) {
        sred[threadIdx.x] = sred[threadIdx.x + off];
        sidx[threadIdx.x] = sidx[threadIdx.x + off];
      }
      __syncthreads();
    }
    if (threadIdx.x == 0) cur = sidx[0];
    __syncthreads();
  }
}

__global__ void k_gather3(const float* __restrict__ xyz, const int* __restrict__ idx,
                          int Bc, int N, int S, float* __restrict__ out) {
  int t = blockIdx.x * blockDim.x + threadIdx.x;
  if (t >= Bc * S) return;
  int b = t / S;
  int n = idx[t];
  const float* s = xyz + ((size_t)b * N + n) * 3;
  float* d = out + (size_t)t * 3;
  d[0] = s[0]; d[1] = s[1]; d[2] = s[2];
}

// Ball query: first <=32 indices within radius (== sorted-index semantics).
__global__ void k_ball(const float* __restrict__ xyz, const float* __restrict__ nxyz,
                       int Bc, int N, int S, float r2, int* __restrict__ out) {
  int t = blockIdx.x * blockDim.x + threadIdx.x;
  if (t >= Bc * S) return;
  int b = t / S;
  const float* q = nxyz + (size_t)t * 3;
  const float* p = xyz + (size_t)b * N * 3;
  int* o = out + (size_t)t * 32;
  int cnt = 0, first = 0;
  bool found = false;
  for (int n = 0; n < N && cnt < 32; ++n) {
    float dx = p[n * 3 + 0] - q[0];
    float dy = p[n * 3 + 1] - q[1];
    float dz = p[n * 3 + 2] - q[2];
    if (dx * dx + dy * dy + dz * dz <= r2) {
      if (!found) { found = true; first = n; }
      o[cnt++] = n;
    }
  }
  for (; cnt < 32; ++cnt) o[cnt] = first;
}

// Grouped rows (f16): [xyz - center | feats | zero pad to Cinp] per (b,s,k).
__global__ void k_group(const float* __restrict__ xyz, const float* __restrict__ nxyz,
                        const _Float16* __restrict__ feats, int Cf, int Cinp,
                        const int* __restrict__ idx, int Bc, int N, int S,
                        _Float16* __restrict__ g) {
  int t = blockIdx.x * blockDim.x + threadIdx.x;
  int rows = Bc * S * 32;
  if (t >= rows) return;
  int ps = t >> 5;          // b*S + s
  int b = ps / S;
  int n = idx[t];
  _Float16* dst = g + (size_t)t * Cinp;
  const float* pn = xyz + ((size_t)b * N + n) * 3;
  const float* q = nxyz + (size_t)ps * 3;
  dst[0] = (_Float16)(pn[0] - q[0]);
  dst[1] = (_Float16)(pn[1] - q[1]);
  dst[2] = (_Float16)(pn[2] - q[2]);
  if (Cf > 0) {
    const _Float16* f = feats + ((size_t)b * N + n) * Cf;
    for (int c = 0; c < Cf; ++c) dst[3 + c] = f[c];
  }
  for (int c = 3 + Cf; c < Cinp; ++c) dst[c] = (_Float16)0.0f;
}

__global__ void k_maxpool32(const _Float16* __restrict__ in, int SB, int C,
                            _Float16* __restrict__ out) {
  int t = blockIdx.x * blockDim.x + threadIdx.x;
  if (t >= SB * C) return;
  int p = t / C, c = t % C;
  const _Float16* base = in + ((size_t)p * 32) * C + c;
  float m = -1e30f;
#pragma unroll 4
  for (int k = 0; k < 32; ++k) m = fmaxf(m, (float)base[(size_t)k * C]);
  out[t] = (_Float16)m;
}

__global__ void k_knn3(const float* __restrict__ xyz1, const float* __restrict__ xyz2,
                       int Bc, int N1, int N2,
                       int* __restrict__ idx3, float* __restrict__ w3) {
  int t = blockIdx.x * blockDim.x + threadIdx.x;
  if (t >= Bc * N1) return;
  int b = t / N1;
  const float* q = xyz1 + (size_t)t * 3;
  const float* p = xyz2 + (size_t)b * N2 * 3;
  float d0 = 1e30f, d1 = 1e30f, d2 = 1e30f;
  int i0 = 0, i1 = 0, i2 = 0;
  for (int m = 0; m < N2; ++m) {
    float dx = q[0] - p[m * 3 + 0];
    float dy = q[1] - p[m * 3 + 1];
    float dz = q[2] - p[m * 3 + 2];
    float d = dx * dx + dy * dy + dz * dz;
    if (d < d0)      { d2 = d1; i2 = i1; d1 = d0; i1 = i0; d0 = d; i0 = m; }
    else if (d < d1) { d2 = d1; i2 = i1; d1 = d; i1 = m; }
    else if (d < d2) { d2 = d; i2 = m; }
  }
  float w0 = 1.0f / fmaxf(d0, 1e-10f);
  float w1 = 1.0f / fmaxf(d1, 1e-10f);
  float w2 = 1.0f / fmaxf(d2, 1e-10f);
  float s = w0 + w1 + w2;
  idx3[(size_t)t * 3 + 0] = i0; idx3[(size_t)t * 3 + 1] = i1; idx3[(size_t)t * 3 + 2] = i2;
  w3[(size_t)t * 3 + 0] = w0 / s; w3[(size_t)t * 3 + 1] = w1 / s; w3[(size_t)t * 3 + 2] = w2 / s;
}

// out row (f16) = [p1 row (C1) | sum_k w_k * p2[idx_k] (C2)]
__global__ void k_interp_concat(const _Float16* __restrict__ p1, int C1,
                                const _Float16* __restrict__ p2, int C2,
                                const int* __restrict__ idx3, const float* __restrict__ w3,
                                int Bc, int N1, int N2, _Float16* __restrict__ out) {
  int t = blockIdx.x * blockDim.x + threadIdx.x;
  if (t >= Bc * N1) return;
  int b = t / N1;
  const int* id = idx3 + (size_t)t * 3;
  const float* w = w3 + (size_t)t * 3;
  _Float16* dst = out + (size_t)t * (C1 + C2);
  if (p1) {
    const _Float16* src = p1 + (size_t)t * C1;
    for (int c = 0; c < C1; ++c) dst[c] = src[c];
  }
  const _Float16* base = p2 + (size_t)b * N2 * C2;
  const _Float16* r0 = base + (size_t)id[0] * C2;
  const _Float16* r1 = base + (size_t)id[1] * C2;
  const _Float16* r2 = base + (size_t)id[2] * C2;
  for (int c = 0; c < C2; ++c)
    dst[C1 + c] = (_Float16)(w[0] * (float)r0[c] + w[1] * (float)r1[c] + w[2] * (float)r2[c]);
}

__global__ __launch_bounds__(THREADS)
void k_bn_stats(const _Float16* __restrict__ x, int M, int C, float* __restrict__ stats) {
  int c = blockIdx.x;
  __shared__ float s1[THREADS], s2[THREADS];
  float a = 0.0f, b = 0.0f;
  for (int i = threadIdx.x; i < M; i += THREADS) {
    float v = (float)x[(size_t)i * C + c];
    a += v; b += v * v;
  }
  s1[threadIdx.x] = a; s2[threadIdx.x] = b;
  __syncthreads();
  for (int off = THREADS / 2; off > 0; off >>= 1) {
    if (threadIdx.x < off) {
      s1[threadIdx.x] += s1[threadIdx.x + off];
      s2[threadIdx.x] += s2[threadIdx.x + off];
    }
    __syncthreads();
  }
  if (threadIdx.x == 0) {
    float mu = s1[0] / (float)M;
    float var = s2[0] / (float)M - mu * mu;
    stats[2 * c + 0] = mu;
    stats[2 * c + 1] = rsqrtf(var + 1e-5f);
  }
}

__global__ void k_bn_apply_relu(_Float16* __restrict__ x, int M, int C,
                                const float* __restrict__ stats,
                                const float* __restrict__ g,
                                const float* __restrict__ beta) {
  int t = blockIdx.x * blockDim.x + threadIdx.x;
  if (t >= M * C) return;
  int c = t % C;
  float v = ((float)x[t] - stats[2 * c]) * stats[2 * c + 1] * g[c] + beta[c];
  x[t] = (_Float16)fmaxf(v, 0.0f);
}

__global__ void k_head_tail(const _Float16* __restrict__ x, int M, int K,
                            const float* __restrict__ W, const float* __restrict__ b,
                            float* __restrict__ out, int Cout) {
  int t = blockIdx.x * blockDim.x + threadIdx.x;
  if (t >= M * Cout) return;
  int m = t / Cout, o = t % Cout;
  const _Float16* xr = x + (size_t)m * K;
  const float* wr = W + (size_t)o * K;
  float s = b[o];
  for (int c = 0; c < K; ++c) s += (float)xr[c] * wr[c];
  out[t] = s;
}

__global__ void k_sigmoid(const float* __restrict__ in, float* __restrict__ out, int n) {
  int t = blockIdx.x * blockDim.x + threadIdx.x;
  if (t >= n) return;
  out[t] = 1.0f / (1.0f + expf(-in[t]));
}

__global__ void k_decode(const float* __restrict__ box, int Bc, int N,
                         float* __restrict__ out) {
  int t = blockIdx.x * blockDim.x + threadIdx.x;
  if (t >= Bc * N) return;
  const float* bp = box + (size_t)t * 7;
  float cx = bp[0], cy = bp[1], cz = bp[2];
  float dx = expf(bp[3]) * 0.5f, dy = expf(bp[4]) * 0.5f, dz = expf(bp[5]) * 0.5f;
  float h = bp[6];
  float ch = cosf(h), sh = sinf(h);
  int b = t / N, n = t % N;
  float* dst = out + ((size_t)b * N * 8 + (size_t)n * 8) * 3;
#pragma unroll
  for (int j = 0; j < 8; ++j) {
    float ux = (j & 4) ? 1.0f : -1.0f;
    float uy = (j & 2) ? 1.0f : -1.0f;
    float uz = (j & 1) ? 1.0f : -1.0f;
    float lx = ux * dx, ly = uy * dy, lz = uz * dz;
    dst[j * 3 + 0] = lx * ch - ly * sh + cx;
    dst[j * 3 + 1] = lx * sh + ly * ch + cy;
    dst[j * 3 + 2] = lz + cz;
  }
}

// ---------------------------------------------------------------------------
// Host orchestration
// ---------------------------------------------------------------------------

// Input pointer order = JAX pytree flatten (sorted dict keys, list/tuple order).
enum {
  BOX_BN_BETA = 0, BOX_BN_G, BOX_C1_B, BOX_C1_W, BOX_C2_B, BOX_C2_W,
  CLS_BN_BETA, CLS_BN_G, CLS_C1_B, CLS_C1_W, CLS_C2_B, CLS_C2_W,
  FP1_0_B, FP1_0_W, FP1_0_BETA, FP1_0_G,
  FP1_1_B, FP1_1_W, FP1_1_BETA, FP1_1_G,
  FP1_2_B, FP1_2_W, FP1_2_BETA, FP1_2_G,
  FP2_0_B, FP2_0_W, FP2_0_BETA, FP2_0_G,
  FP2_1_B, FP2_1_W, FP2_1_BETA, FP2_1_G,
  FP3_0_B, FP3_0_W, FP3_0_BETA, FP3_0_G,
  FP3_1_B, FP3_1_W, FP3_1_BETA, FP3_1_G,
  SA1_0_B, SA1_0_W, SA1_1_B, SA1_1_W, SA1_2_B, SA1_2_W,
  SA2_0_B, SA2_0_W, SA2_1_B, SA2_1_W, SA2_2_B, SA2_2_W,
  SA3_0_B, SA3_0_W, SA3_1_B, SA3_1_W, SA3_2_B, SA3_2_W,
  POINTS, N_INPUTS
};

extern "C" void kernel_launch(void* const* d_in, const int* in_sizes, int n_in,
                              void* d_out, int out_size, void* d_ws, size_t ws_size,
                              hipStream_t stream) {
  (void)in_sizes; (void)n_in; (void)out_size; (void)ws_size;
  const float* P[N_INPUTS];
  for (int i = 0; i < N_INPUTS; ++i) P[i] = (const float*)d_in[i];

  const int B = 4, N0 = 4096;
  const int S1 = 2048, S2 = 512, S3 = 128;

  // ---- workspace carving -------------------------------------------------
  char* ws = (char*)d_ws;
  size_t cur = 0;
  auto alloc = [&](size_t bytes) -> void* {
    void* p = ws + cur;
    cur += (bytes + 255) & ~(size_t)255;
    return p;
  };
  float*     dist    = (float*)    alloc((size_t)B * N0 * 4);
  int*       fidx1   = (int*)      alloc((size_t)B * S1 * 4);
  int*       fidx2   = (int*)      alloc((size_t)B * S2 * 4);
  int*       fidx3   = (int*)      alloc((size_t)B * S3 * 4);
  float*     xyz1    = (float*)    alloc((size_t)B * S1 * 3 * 4);
  float*     xyz2    = (float*)    alloc((size_t)B * S2 * 3 * 4);
  float*     xyz3    = (float*)    alloc((size_t)B * S3 * 3 * 4);
  int*       ballidx = (int*)      alloc((size_t)B * S1 * 32 * 4);
  int*       kidx    = (int*)      alloc((size_t)B * N0 * 3 * 4);
  float*     kw      = (float*)    alloc((size_t)B * N0 * 3 * 4);
  float*     stats   = (float*)    alloc((size_t)1024 * 4);
  _Float16*  f1      = (_Float16*) alloc((size_t)B * S1 * 128 * 2);
  _Float16*  f2      = (_Float16*) alloc((size_t)B * S2 * 256 * 2);
  _Float16*  f3      = (_Float16*) alloc((size_t)B * S3 * 512 * 2);
  float*     clsbuf  = (float*)    alloc((size_t)B * N0 * 4);
  float*     boxbuf  = (float*)    alloc((size_t)B * N0 * 7 * 4);
  _Float16*  gbuf    = (_Float16*) alloc((size_t)16 * 1024 * 1024 * 2);
  _Float16*  ping    = (_Float16*) alloc((size_t)262144 * 128 * 2);
  _Float16*  pong    = (_Float16*) alloc((size_t)262144 * 128 * 2);

  auto blocks = [](int n) { return (n + THREADS - 1) / THREADS; };

  // Pre-convert all weights to zero-padded f16 (Kp = K rounded up to 32).
  auto prep_w = [&](int widx, int Cout, int Kreal) -> _Float16* {
    int Kp = (Kreal + 31) & ~31;
    _Float16* wh = (_Float16*)alloc((size_t)Cout * Kp * 2);
    k_w2h<<<blocks(Cout * Kp), THREADS, 0, stream>>>(P[widx], Cout, Kreal, Kp, wh);
    return wh;
  };
  _Float16* Wsa1_0 = prep_w(SA1_0_W,  64,   3);
  _Float16* Wsa1_1 = prep_w(SA1_1_W,  64,  64);
  _Float16* Wsa1_2 = prep_w(SA1_2_W, 128,  64);
  _Float16* Wsa2_0 = prep_w(SA2_0_W, 128, 131);
  _Float16* Wsa2_1 = prep_w(SA2_1_W, 128, 128);
  _Float16* Wsa2_2 = prep_w(SA2_2_W, 256, 128);
  _Float16* Wsa3_0 = prep_w(SA3_0_W, 256, 259);
  _Float16* Wsa3_1 = prep_w(SA3_1_W, 256, 256);
  _Float16* Wsa3_2 = prep_w(SA3_2_W, 512, 256);
  _Float16* Wfp3_0 = prep_w(FP3_0_W, 256, 768);
  _Float16* Wfp3_1 = prep_w(FP3_1_W, 256, 256);
  _Float16* Wfp2_0 = prep_w(FP2_0_W, 256, 384);
  _Float16* Wfp2_1 = prep_w(FP2_1_W, 128, 256);
  _Float16* Wfp1_0 = prep_w(FP1_0_W, 128, 128);
  _Float16* Wfp1_1 = prep_w(FP1_1_W, 128, 128);
  _Float16* Wfp1_2 = prep_w(FP1_2_W, 128, 128);
  _Float16* Wcls1  = prep_w(CLS_C1_W, 128, 128);
  _Float16* Wbox1  = prep_w(BOX_C1_W, 128, 128);

  auto gemm = [&](const _Float16* A, int M, int Kp, const _Float16* Wh,
                  const float* b, _Float16* out, int Cout, int act) {
    dim3 grid(Cout / 64, M / 64);
    k_gemm_wmma_f16<<<grid, THREADS, 0, stream>>>(A, M, Kp, Wh, b, out, Cout, act);
  };
  auto bn = [&](_Float16* x, int M, int C, const float* g, const float* beta) {
    k_bn_stats<<<C, THREADS, 0, stream>>>(x, M, C, stats);
    k_bn_apply_relu<<<blocks(M * C), THREADS, 0, stream>>>(x, M, C, stats, g, beta);
  };

  const float* points = P[POINTS];

  // ---- SA1: 4096 -> 2048, r=0.1, (3 pad 32) -> 64 -> 64 -> 128 ------------
  k_fps<<<B, THREADS, 0, stream>>>(points, N0, S1, dist, fidx1);
  k_gather3<<<blocks(B * S1), THREADS, 0, stream>>>(points, fidx1, B, N0, S1, xyz1);
  k_ball<<<blocks(B * S1), THREADS, 0, stream>>>(points, xyz1, B, N0, S1, 0.01f, ballidx);
  {
    int rows = B * S1 * 32;                                     // 262144
    k_group<<<blocks(rows), THREADS, 0, stream>>>(points, xyz1, nullptr, 0, 32,
                                                  ballidx, B, N0, S1, gbuf);
    gemm(gbuf, rows,  32, Wsa1_0, P[SA1_0_B], ping, 64, 1);
    gemm(ping, rows,  64, Wsa1_1, P[SA1_1_B], pong, 64, 1);
    gemm(pong, rows,  64, Wsa1_2, P[SA1_2_B], ping, 128, 0);
    k_maxpool32<<<blocks(B * S1 * 128), THREADS, 0, stream>>>(ping, B * S1, 128, f1);
  }

  // ---- SA2: 2048 -> 512, r=0.2, (131 pad 160) -> 128 -> 128 -> 256 --------
  k_fps<<<B, THREADS, 0, stream>>>(xyz1, S1, S2, dist, fidx2);
  k_gather3<<<blocks(B * S2), THREADS, 0, stream>>>(xyz1, fidx2, B, S1, S2, xyz2);
  k_ball<<<blocks(B * S2), THREADS, 0, stream>>>(xyz1, xyz2, B, S1, S2, 0.04f, ballidx);
  {
    int rows = B * S2 * 32;                                     // 65536
    k_group<<<blocks(rows), THREADS, 0, stream>>>(xyz1, xyz2, f1, 128, 160,
                                                  ballidx, B, S1, S2, gbuf);
    gemm(gbuf, rows, 160, Wsa2_0, P[SA2_0_B], ping, 128, 1);
    gemm(ping, rows, 128, Wsa2_1, P[SA2_1_B], pong, 128, 1);
    gemm(pong, rows, 128, Wsa2_2, P[SA2_2_B], ping, 256, 0);
    k_maxpool32<<<blocks(B * S2 * 256), THREADS, 0, stream>>>(ping, B * S2, 256, f2);
  }

  // ---- SA3: 512 -> 128, r=0.4, (259 pad 288) -> 256 -> 256 -> 512 ---------
  k_fps<<<B, THREADS, 0, stream>>>(xyz2, S2, S3, dist, fidx3);
  k_gather3<<<blocks(B * S3), THREADS, 0, stream>>>(xyz2, fidx3, B, S2, S3, xyz3);
  k_ball<<<blocks(B * S3), THREADS, 0, stream>>>(xyz2, xyz3, B, S2, S3, 0.16f, ballidx);
  {
    int rows = B * S3 * 32;                                     // 16384
    k_group<<<blocks(rows), THREADS, 0, stream>>>(xyz2, xyz3, f2, 256, 288,
                                                  ballidx, B, S2, S3, gbuf);
    gemm(gbuf, rows, 288, Wsa3_0, P[SA3_0_B], ping, 256, 1);
    gemm(ping, rows, 256, Wsa3_1, P[SA3_1_B], pong, 256, 1);
    gemm(pong, rows, 256, Wsa3_2, P[SA3_2_B], ping, 512, 0);
    k_maxpool32<<<blocks(B * S3 * 512), THREADS, 0, stream>>>(ping, B * S3, 512, f3);
  }

  // ---- FP3: interp f3 (128 pts) onto xyz2 (512 pts), concat f2 ------------
  {
    int M = B * S2;                                             // 2048
    k_knn3<<<blocks(M), THREADS, 0, stream>>>(xyz2, xyz3, B, S2, S3, kidx, kw);
    k_interp_concat<<<blocks(M), THREADS, 0, stream>>>(f2, 256, f3, 512,
                                                       kidx, kw, B, S2, S3, gbuf);
    gemm(gbuf, M, 768, Wfp3_0, P[FP3_0_B], ping, 256, 0);
    bn(ping, M, 256, P[FP3_0_G], P[FP3_0_BETA]);
    gemm(ping, M, 256, Wfp3_1, P[FP3_1_B], pong, 256, 0);
    bn(pong, M, 256, P[FP3_1_G], P[FP3_1_BETA]);                // new f2 in pong
  }

  // ---- FP2: interp onto xyz1 (2048 pts), concat f1 ------------------------
  {
    int M = B * S1;                                             // 8192
    k_knn3<<<blocks(M), THREADS, 0, stream>>>(xyz1, xyz2, B, S1, S2, kidx, kw);
    k_interp_concat<<<blocks(M), THREADS, 0, stream>>>(f1, 128, pong, 256,
                                                       kidx, kw, B, S1, S2, gbuf);
    gemm(gbuf, M, 384, Wfp2_0, P[FP2_0_B], ping, 256, 0);
    bn(ping, M, 256, P[FP2_0_G], P[FP2_0_BETA]);
    gemm(ping, M, 256, Wfp2_1, P[FP2_1_B], pong, 128, 0);
    bn(pong, M, 128, P[FP2_1_G], P[FP2_1_BETA]);                // new f1 in pong
  }

  // ---- FP1: interp onto original xyz (4096 pts), no concat ----------------
  {
    int M = B * N0;                                             // 16384
    k_knn3<<<blocks(M), THREADS, 0, stream>>>(points, xyz1, B, N0, S1, kidx, kw);
    k_interp_concat<<<blocks(M), THREADS, 0, stream>>>(nullptr, 0, pong, 128,
                                                       kidx, kw, B, N0, S1, gbuf);
    gemm(gbuf, M, 128, Wfp1_0, P[FP1_0_B], ping, 128, 0);
    bn(ping, M, 128, P[FP1_0_G], P[FP1_0_BETA]);
    gemm(ping, M, 128, Wfp1_1, P[FP1_1_B], pong, 128, 0);
    bn(pong, M, 128, P[FP1_1_G], P[FP1_1_BETA]);
    gemm(pong, M, 128, Wfp1_2, P[FP1_2_B], ping, 128, 0);
    bn(ping, M, 128, P[FP1_2_G], P[FP1_2_BETA]);                // f0 in ping
  }

  // ---- heads --------------------------------------------------------------
  float* out = (float*)d_out;
  {
    int M = B * N0;                                             // 16384
    // cls head
    gemm(ping, M, 128, Wcls1, P[CLS_C1_B], pong, 128, 0);
    bn(pong, M, 128, P[CLS_BN_G], P[CLS_BN_BETA]);
    k_head_tail<<<blocks(M * 1), THREADS, 0, stream>>>(pong, M, 128,
                                                       P[CLS_C2_W], P[CLS_C2_B], clsbuf, 1);
    k_sigmoid<<<blocks(M), THREADS, 0, stream>>>(clsbuf, out, M);
    // box head
    gemm(ping, M, 128, Wbox1, P[BOX_C1_B], pong, 128, 0);
    bn(pong, M, 128, P[BOX_BN_G], P[BOX_BN_BETA]);
    k_head_tail<<<blocks(M * 7), THREADS, 0, stream>>>(pong, M, 128,
                                                       P[BOX_C2_W], P[BOX_C2_B], boxbuf, 7);
    k_decode<<<blocks(M), THREADS, 0, stream>>>(boxbuf, B, N0, out + M);
  }
}
